// SelfAttention2D_5600637354162
// MI455X (gfx1250) — matrix operand order repaired and checked by reading, not compiled, not run on hardware
//
#include <hip/hip_runtime.h>
#include <hip/hip_bf16.h>

#define DEV __device__ __forceinline__

typedef __attribute__((ext_vector_type(16))) __bf16 bf16x16;
typedef __attribute__((ext_vector_type(8)))  __bf16 bf16x8;
typedef __attribute__((ext_vector_type(8)))  float  f32x8;
typedef __attribute__((ext_vector_type(4)))  unsigned u32x4;
typedef __attribute__((ext_vector_type(4)))  int      i32x4;

static constexpr int C     = 256;
static constexpr int NPIX  = 4096;  // 64*64 pixels
static constexpr int NB    = 8;     // batch
static constexpr int ROWS1 = 128;   // pixel rows per block (qkv / attn)
static constexpr int KTILE = 64;    // key rows per attention tile
static constexpr int ROWS3 = 64;    // pixel rows per block (proj)
static constexpr int LDA   = 264;   // LDS stride, 256-wide bf16 tiles (mult of 8)
static constexpr int LDK   = 72;    // LDS stride, 64-wide bf16 tiles (mult of 8)
static constexpr int LDT   = 136;   // LDS stride, 128-wide bf16 tiles (mult of 8)
static constexpr int LDO   = 68;    // LDS stride, 64-wide f32 tiles

// ---------------------------------------------------------------------------
// CDNA5 async global->LDS copies (ASYNCcnt), guarded; sync vector fallback.
// Builtin signature (per clang diagnostic): (v4i AS1*, v4i AS3*, Iimm, Iimm).
// ---------------------------------------------------------------------------
#if defined(__gfx1250__) && __has_builtin(__builtin_amdgcn_global_load_async_to_lds_b128)
#define HAS_ASYNC 1
typedef __attribute__((address_space(1))) i32x4 gi32x4;
typedef __attribute__((address_space(3))) i32x4 li32x4;
DEV void cp16(const void* g, void* l) {
  // flat->AS3 is a 32-bit truncation on AMDGPU; flat->AS1 is identity.
  __builtin_amdgcn_global_load_async_to_lds_b128(
      (gi32x4*)(unsigned long long)(size_t)g,
      (li32x4*)(unsigned)(size_t)l, 0, 0);
}
DEV void async_fence() {
#if __has_builtin(__builtin_amdgcn_s_wait_asynccnt)
  __builtin_amdgcn_s_wait_asynccnt(0);
#else
  asm volatile("s_wait_asynccnt 0x0" ::: "memory");
#endif
}
#else
#define HAS_ASYNC 0
DEV void cp16(const void* g, void* l) {
  *(u32x4*)l = *(const u32x4*)g;
}
DEV void async_fence() {}
#endif

DEV int lane_id() { return threadIdx.x & 31; }

// A fragment: 16x32 (MxK) bf16 from row-major [M][K] storage.
DEV bf16x16 load_frag_a(const __bf16* p, int ld) {
  int lane = lane_id();
  int h = lane >> 4, m = lane & 15;
  const __bf16* r = p + m * ld + h * 8;
  bf16x8 lo = __builtin_bit_cast(bf16x8, *(const u32x4*)(r));
  bf16x8 hi = __builtin_bit_cast(bf16x8, *(const u32x4*)(r + 16));
  bf16x16 a;
#pragma unroll
  for (int i = 0; i < 8; ++i) { a[i] = lo[i]; a[i + 8] = hi[i]; }
  return a;
}

// B fragment: 32x16 (KxN) bf16, source stored row-major as [N][K] (i.e. B^T).
DEV bf16x16 load_frag_b(const __bf16* p, int ld) {
  int lane = lane_id();
  int h = lane >> 4, n = lane & 15;
  const __bf16* r = p + n * ld + h * 8;
  bf16x8 lo = __builtin_bit_cast(bf16x8, *(const u32x4*)(r));
  bf16x8 hi = __builtin_bit_cast(bf16x8, *(const u32x4*)(r + 16));
  bf16x16 b;
#pragma unroll
  for (int i = 0; i < 8; ++i) { b[i] = lo[i]; b[i + 8] = hi[i]; }
  return b;
}

DEV f32x8 wmma_bf16(bf16x16 a, bf16x16 b, f32x8 c) {
  return __builtin_amdgcn_wmma_f32_16x16x32_bf16(
      false, a, false, b, (short)0, c, false, false);
}

// ---------------------------------------------------------------------------
// Kernel 0: convert the four 256x256 f32 weight matrices to bf16 in workspace.
// ---------------------------------------------------------------------------
__global__ void wcvt_kernel(const float* __restrict__ Wq, const float* __restrict__ Wk,
                            const float* __restrict__ Wv, const float* __restrict__ Wp,
                            __bf16* __restrict__ out) {
  int i = blockIdx.x * blockDim.x + threadIdx.x;
  if (i < C * C) {
    out[0 * C * C + i] = (__bf16)Wq[i];
    out[1 * C * C + i] = (__bf16)Wk[i];
    out[2 * C * C + i] = (__bf16)Wv[i];
    out[3 * C * C + i] = (__bf16)Wp[i];
  }
}

// ---------------------------------------------------------------------------
// Kernel 1: Q/K/V projection. 128 pixels x 256 channels per block, 8 waves.
// Emits Q,K as bf16 [B][N][C] and V *transposed* as bf16 [B][C][N] so the
// attention kernel needs no transpose in its hot loop.
// ---------------------------------------------------------------------------
__global__ void qkv_kernel(const float* __restrict__ x,       // [B][C][N]
                           const __bf16* __restrict__ Wb,     // [3][C][C] bf16
                           const float* __restrict__ bq,
                           const float* __restrict__ bk,
                           const float* __restrict__ bv,
                           __bf16* __restrict__ qbuf,         // [B][N][C]
                           __bf16* __restrict__ kbuf,         // [B][N][C]
                           __bf16* __restrict__ vtbuf)        // [B][C][N]
{
  extern __shared__ __align__(16) char smem[];
  __bf16* Xs = (__bf16*)smem;           // [ROWS1][LDA]
  __bf16* Ws = Xs + ROWS1 * LDA;        // [C][LDA]
  __bf16* Ts = Ws + C * LDA;            // staging: [ROWS1][LDA] or [C][LDT]
  const int n0   = blockIdx.x * ROWS1;
  const int b    = blockIdx.y;
  const int tid  = threadIdx.x;
  const int wave = tid >> 5;
  const int lane = tid & 31;
  const int h = lane >> 4, nl = lane & 15;
  const int m0 = wave * 16;

  // transpose-load x tile: [C][N] -> Xs[pixel][channel]; float4 global reads
  for (int idx = tid; idx < ROWS1 * C / 4; idx += blockDim.x) {
    int c  = idx >> 5;
    int n4 = (idx & 31) << 2;
    float4 v = *(const float4*)(x + ((size_t)b * C + c) * NPIX + n0 + n4);
    Xs[(n4 + 0) * LDA + c] = (__bf16)v.x;
    Xs[(n4 + 1) * LDA + c] = (__bf16)v.y;
    Xs[(n4 + 2) * LDA + c] = (__bf16)v.z;
    Xs[(n4 + 3) * LDA + c] = (__bf16)v.w;
  }
  __syncthreads();

  // A fragments resident across all 3 matrices and all 16 output tiles
  bf16x16 ax[8];
#pragma unroll
  for (int ks = 0; ks < 8; ++ks) ax[ks] = load_frag_a(&Xs[m0 * LDA + ks * 32], LDA);

  const float* biases[3] = {bq, bk, bv};
  for (int mtx = 0; mtx < 3; ++mtx) {
    __syncthreads();
    // stage bf16 weight matrix into LDS (16-byte chunks)
    const __bf16* wsrc = Wb + (size_t)mtx * C * C;
    for (int idx = tid; idx < (C * C) / 8; idx += blockDim.x) {
      int o  = idx >> 5;
      int c8 = (idx & 31) << 3;
      *(u32x4*)&Ws[o * LDA + c8] = *(const u32x4*)(wsrc + o * C + c8);
    }
    __syncthreads();

    const float* bias = biases[mtx];
    for (int ct = 0; ct < 16; ++ct) {
      f32x8 acc;
      float bval = bias[ct * 16 + nl];
#pragma unroll
      for (int r = 0; r < 8; ++r) acc[r] = bval;
#pragma unroll
      for (int ks = 0; ks < 8; ++ks) {
        bf16x16 bb = load_frag_b(&Ws[(ct * 16) * LDA + ks * 32], LDA);
        acc = wmma_bf16(ax[ks], bb, acc);
      }
      if (mtx < 2) {       // stage [pixel][chan]
#pragma unroll
        for (int r = 0; r < 8; ++r)
          Ts[(m0 + r + 8 * h) * LDA + ct * 16 + nl] = (__bf16)acc[r];
      } else {             // V: stage transposed [chan][pixel]
#pragma unroll
        for (int r = 0; r < 8; ++r)
          Ts[(ct * 16 + nl) * LDT + m0 + r + 8 * h] = (__bf16)acc[r];
      }
    }
    __syncthreads();
    if (mtx < 2) {
      __bf16* dst = (mtx ? kbuf : qbuf) + ((size_t)b * NPIX + n0) * C;
      for (int idx = tid; idx < ROWS1 * C / 8; idx += blockDim.x) {
        int nn = idx >> 5;
        int c8 = (idx & 31) << 3;
        *(u32x4*)(dst + (size_t)nn * C + c8) = *(const u32x4*)&Ts[nn * LDA + c8];
      }
    } else {
      for (int idx = tid; idx < C * ROWS1 / 8; idx += blockDim.x) {
        int c  = idx >> 4;
        int n8 = (idx & 15) << 3;
        *(u32x4*)(vtbuf + ((size_t)b * C + c) * NPIX + n0 + n8) =
            *(const u32x4*)&Ts[c * LDT + n8];
      }
    }
  }
}

// ---------------------------------------------------------------------------
// Kernel 2: flash attention. 128 query rows per block (8 waves, 16 rows/wave),
// double-buffered 64-key K/V^T tiles fed by async global->LDS b128 copies.
// ---------------------------------------------------------------------------
DEV void stage_tile(const __bf16* kg, const __bf16* vtg, int kv,
                    __bf16* Kd, __bf16* Vd, int tid) {
  // K tile: [KTILE][C] -> Kd[KTILE][LDA]
#pragma unroll
  for (int j = 0; j < 8; ++j) {
    int idx = tid + j * 256;
    int kk = idx >> 5, c8 = (idx & 31) << 3;
    cp16(kg + (size_t)(kv + kk) * C + c8, &Kd[kk * LDA + c8]);
  }
  // V^T tile: [C][KTILE] slice of [C][NPIX] -> Vd[C][LDK]
#pragma unroll
  for (int j = 0; j < 8; ++j) {
    int idx = tid + j * 256;
    int c = idx >> 3, n8 = (idx & 7) << 3;
    cp16(vtg + (size_t)c * NPIX + kv + n8, &Vd[c * LDK + n8]);
  }
}

__global__ void attn_kernel(const __bf16* __restrict__ qbuf,  // [B][N][C]
                            const __bf16* __restrict__ kbuf,  // [B][N][C]
                            const __bf16* __restrict__ vtbuf, // [B][C][N]
                            __bf16* __restrict__ obuf)        // [B][N][C]
{
  extern __shared__ __align__(16) char smem[];
  __bf16* Qs  = (__bf16*)smem;           // [ROWS1][LDA]
  __bf16* Ps  = Qs + ROWS1 * LDA;        // [ROWS1][LDK]
  __bf16* KsB = Ps + ROWS1 * LDK;        // [2][KTILE][LDA]
  __bf16* VtB = KsB + 2 * KTILE * LDA;   // [2][C][LDK]

  const int n0   = blockIdx.x * ROWS1;
  const int b    = blockIdx.y;
  const int tid  = threadIdx.x;
  const int wave = tid >> 5;
  const int lane = tid & 31;
  const int h = lane >> 4, nl = lane & 15;
  const int m0 = wave * 16;

  const __bf16* qg  = qbuf  + (size_t)b * NPIX * C;
  const __bf16* kg  = kbuf  + (size_t)b * NPIX * C;
  const __bf16* vtg = vtbuf + (size_t)b * C * NPIX;

  stage_tile(kg, vtg, 0, KsB, VtB, tid);   // tile 0 in flight

  for (int idx = tid; idx < ROWS1 * C / 8; idx += blockDim.x) {
    int nn = idx >> 5, c8 = (idx & 31) << 3;
    *(u32x4*)&Qs[nn * LDA + c8] = *(const u32x4*)(qg + (size_t)(n0 + nn) * C + c8);
  }
  __syncthreads();

  bf16x16 aq[8];
#pragma unroll
  for (int ks = 0; ks < 8; ++ks) aq[ks] = load_frag_a(&Qs[m0 * LDA + ks * 32], LDA);

  f32x8 oacc[16];
#pragma unroll
  for (int t = 0; t < 16; ++t)
#pragma unroll
    for (int r = 0; r < 8; ++r) oacc[t][r] = 0.f;
  float mrow[8], lrow[8];
#pragma unroll
  for (int r = 0; r < 8; ++r) { mrow[r] = -3.0e38f; lrow[r] = 0.f; }

  const float scale = 0.0625f;   // C^-0.5
  const int NT = NPIX / KTILE;

  for (int it = 0; it < NT; ++it) {
    async_fence();     // my tile-`it` async ops complete (in-order)
    __syncthreads();   // all portions landed; all waves done with prior tiles
    __bf16* Kc = KsB + (it & 1) * (KTILE * LDA);
    __bf16* Vc = VtB + (it & 1) * (C * LDK);
    if (it + 1 < NT)
      stage_tile(kg, vtg, (it + 1) * KTILE,
                 KsB + ((it + 1) & 1) * (KTILE * LDA),
                 VtB + ((it + 1) & 1) * (C * LDK), tid);

    // S = Q * K^T
    f32x8 s[4];
#pragma unroll
    for (int nt = 0; nt < 4; ++nt) {
#pragma unroll
      for (int r = 0; r < 8; ++r) s[nt][r] = 0.f;
#pragma unroll
      for (int ks = 0; ks < 8; ++ks) {
        bf16x16 bb = load_frag_b(&Kc[(nt * 16) * LDA + ks * 32], LDA);
        s[nt] = wmma_bf16(aq[ks], bb, s[nt]);
      }
    }

    float curmax[8];
#pragma unroll
    for (int r = 0; r < 8; ++r) {
      float v = fmaxf(fmaxf(s[0][r], s[1][r]), fmaxf(s[2][r], s[3][r]));
      for (int off = 1; off < 16; off <<= 1) v = fmaxf(v, __shfl_xor(v, off, 16));
      curmax[r] = v * scale;
    }
    float alpha[8];
#pragma unroll
    for (int r = 0; r < 8; ++r) {
      float mnew = fmaxf(mrow[r], curmax[r]);
      alpha[r]   = __expf(mrow[r] - mnew);
      mrow[r]    = mnew;
    }
    float rsum[8];
#pragma unroll
    for (int r = 0; r < 8; ++r) rsum[r] = 0.f;
#pragma unroll
    for (int nt = 0; nt < 4; ++nt) {
#pragma unroll
      for (int r = 0; r < 8; ++r) {
        float p = __expf(s[nt][r] * scale - mrow[r]);
        rsum[r] += p;
        Ps[(m0 + r + 8 * h) * LDK + nt * 16 + nl] = (__bf16)p;
      }
    }
#pragma unroll
    for (int r = 0; r < 8; ++r) {
      float v = rsum[r];
      for (int off = 1; off < 16; off <<= 1) v += __shfl_xor(v, off, 16);
      lrow[r] = lrow[r] * alpha[r] + v;
    }
#pragma unroll
    for (int t = 0; t < 16; ++t)
#pragma unroll
      for (int r = 0; r < 8; ++r) oacc[t][r] *= alpha[r];

    __syncthreads();   // Ps visible

    bf16x16 ap0 = load_frag_a(&Ps[m0 * LDK + 0],  LDK);
    bf16x16 ap1 = load_frag_a(&Ps[m0 * LDK + 32], LDK);
#pragma unroll
    for (int t = 0; t < 16; ++t) {
      bf16x16 b0 = load_frag_b(&Vc[(t * 16) * LDK + 0],  LDK);
      bf16x16 b1 = load_frag_b(&Vc[(t * 16) * LDK + 32], LDK);
      oacc[t] = wmma_bf16(ap0, b0, oacc[t]);
      oacc[t] = wmma_bf16(ap1, b1, oacc[t]);
    }
  }

  // epilogue: normalize, stage to LDS (reuse K buffers), coalesced b128 stores
  __bf16* Ot = KsB;    // [ROWS1][LDA] == 2*[KTILE][LDA]
#pragma unroll
  for (int r = 0; r < 8; ++r) lrow[r] = 1.f / lrow[r];
#pragma unroll
  for (int t = 0; t < 16; ++t)
#pragma unroll
    for (int r = 0; r < 8; ++r)
      Ot[(m0 + r + 8 * h) * LDA + t * 16 + nl] = (__bf16)(oacc[t][r] * lrow[r]);
  __syncthreads();
  __bf16* og = obuf + ((size_t)b * NPIX + n0) * C;
  for (int idx = tid; idx < ROWS1 * C / 8; idx += blockDim.x) {
    int nn = idx >> 5, c8 = (idx & 31) << 3;
    *(u32x4*)(og + (size_t)nn * C + c8) = *(const u32x4*)&Ot[nn * LDA + c8];
  }
}

// ---------------------------------------------------------------------------
// Kernel 3: output projection  out = ((O + xf) @ Wp^T + bp), stored [B][C][N].
// ---------------------------------------------------------------------------
__global__ void proj_kernel(const float* __restrict__ x,     // [B][C][N]
                            const __bf16* __restrict__ obuf, // [B][N][C]
                            const __bf16* __restrict__ Wpb,  // [C][C] bf16
                            const float* __restrict__ bp,
                            float* __restrict__ out)         // [B][C][N]
{
  extern __shared__ __align__(16) char smem[];
  __bf16* As = (__bf16*)smem;            // [ROWS3][LDA]
  __bf16* Ws = As + ROWS3 * LDA;         // [C][LDA]
  float*  Os = (float*)(Ws + C * LDA);   // [C][LDO]

  const int n0   = blockIdx.x * ROWS3;
  const int b    = blockIdx.y;
  const int tid  = threadIdx.x;
  const int wave = tid >> 5;
  const int lane = tid & 31;
  const int h = lane >> 4, nl = lane & 15;
  const int m0 = wave * 16;

  for (int idx = tid; idx < ROWS3 * C / 8; idx += blockDim.x) {
    int nn = idx >> 5, c8 = (idx & 31) << 3;
    *(u32x4*)&As[nn * LDA + c8] =
        *(const u32x4*)(obuf + ((size_t)b * NPIX + n0 + nn) * C + c8);
  }
  for (int idx = tid; idx < (C * C) / 8; idx += blockDim.x) {
    int o = idx >> 5, c8 = (idx & 31) << 3;
    *(u32x4*)&Ws[o * LDA + c8] = *(const u32x4*)(Wpb + o * C + c8);
  }
  __syncthreads();
  // As += xf (float4 reads from [C][N])
  for (int idx = tid; idx < ROWS3 * C / 4; idx += blockDim.x) {
    int c = idx >> 4, n4 = (idx & 15) << 2;
    float4 v = *(const float4*)(x + ((size_t)b * C + c) * NPIX + n0 + n4);
    As[(n4 + 0) * LDA + c] = (__bf16)((float)As[(n4 + 0) * LDA + c] + v.x);
    As[(n4 + 1) * LDA + c] = (__bf16)((float)As[(n4 + 1) * LDA + c] + v.y);
    As[(n4 + 2) * LDA + c] = (__bf16)((float)As[(n4 + 2) * LDA + c] + v.z);
    As[(n4 + 3) * LDA + c] = (__bf16)((float)As[(n4 + 3) * LDA + c] + v.w);
  }
  __syncthreads();

  // A fragments resident across all 16 output tiles
  bf16x16 aa[8];
#pragma unroll
  for (int ks = 0; ks < 8; ++ks) aa[ks] = load_frag_a(&As[m0 * LDA + ks * 32], LDA);

  for (int ct = 0; ct < 16; ++ct) {
    f32x8 acc;
    float bval = bp[ct * 16 + nl];
#pragma unroll
    for (int r = 0; r < 8; ++r) acc[r] = bval;
#pragma unroll
    for (int ks = 0; ks < 8; ++ks) {
      bf16x16 bb = load_frag_b(&Ws[(ct * 16) * LDA + ks * 32], LDA);
      acc = wmma_bf16(aa[ks], bb, acc);
    }
#pragma unroll
    for (int r = 0; r < 8; ++r)
      Os[(ct * 16 + nl) * LDO + m0 + r + 8 * h] = acc[r];
  }
  __syncthreads();
  for (int idx = tid; idx < C * ROWS3 / 4; idx += blockDim.x) {
    int c = idx >> 4, n4 = (idx & 15) << 2;
    *(float4*)(out + ((size_t)b * C + c) * NPIX + n0 + n4) =
        *(const float4*)&Os[c * LDO + n4];
  }
}

// ---------------------------------------------------------------------------
extern "C" void kernel_launch(void* const* d_in, const int* in_sizes, int n_in,
                              void* d_out, int out_size, void* d_ws, size_t ws_size,
                              hipStream_t stream) {
  const float* x  = (const float*)d_in[0];
  const float* Wq = (const float*)d_in[1];
  const float* bq = (const float*)d_in[2];
  const float* Wk = (const float*)d_in[3];
  const float* bk = (const float*)d_in[4];
  const float* Wv = (const float*)d_in[5];
  const float* bv = (const float*)d_in[6];
  const float* Wp = (const float*)d_in[7];
  const float* bp = (const float*)d_in[8];
  float* out = (float*)d_out;

  char* ws = (char*)d_ws;
  __bf16* wbf   = (__bf16*)ws;                               // [4][C*C]
  __bf16* qbuf  = (__bf16*)(ws + (size_t)4 * C * C * 2);     // [B][N][C]
  __bf16* kbuf  = qbuf + (size_t)NB * NPIX * C;              // [B][N][C]
  __bf16* vtbuf = kbuf + (size_t)NB * NPIX * C;              // [B][C][N]
  __bf16* obuf  = vtbuf + (size_t)NB * NPIX * C;             // [B][N][C]

  wcvt_kernel<<<dim3((C * C) / 256), dim3(256), 0, stream>>>(Wq, Wk, Wv, Wp, wbf);

  size_t smem1 = (size_t)(ROWS1 * LDA + C * LDA + C * LDT) * sizeof(__bf16);
  qkv_kernel<<<dim3(NPIX / ROWS1, NB), dim3(256), smem1, stream>>>(
      x, wbf, bq, bk, bv, qbuf, kbuf, vtbuf);

  size_t smem2 = (size_t)(ROWS1 * LDA + ROWS1 * LDK +
                          2 * KTILE * LDA + 2 * C * LDK) * sizeof(__bf16);
  attn_kernel<<<dim3(NPIX / ROWS1, NB), dim3(256), smem2, stream>>>(
      qbuf, kbuf, vtbuf, obuf);

  size_t smem3 = (size_t)(ROWS3 * LDA + C * LDA) * sizeof(__bf16)
               + (size_t)(C * LDO) * sizeof(float);
  proj_kernel<<<dim3(NPIX / ROWS3, NB), dim3(128), smem3, stream>>>(
      x, obuf, wbf + (size_t)3 * C * C, bp, out);
}
